// GATE_19086834663561
// MI455X (gfx1250) — matrix operand order repaired and checked by reading, not compiled, not run on hardware
//
#include <hip/hip_runtime.h>
#include <hip/hip_bf16.h>

#define NROWS 6144
#define ATTR 512
#define HID 256
#define EMB 16
#define ALPHA 0.2f
#define NEG_INF (-9e15f)

typedef __attribute__((ext_vector_type(16))) __bf16 v16bf;
typedef __attribute__((ext_vector_type(8)))  float  v8f;
typedef __attribute__((ext_vector_type(4)))  float  f4v;

union BF16x16 { v16bf v; unsigned int u[8]; uint4 q[2]; };

// f32 -> bf16 via language cast: lowers to native v_cvt_pk_bf16_f32.
__device__ __forceinline__ unsigned int pack2bf(float a, float b) {
    union { __bf16 h[2]; unsigned int u; } t;
    t.h[0] = (__bf16)a;
    t.h[1] = (__bf16)b;
    return t.u;
}

// ---------------------------------------------------------------------------
// Generic C = A(f32)@B(f32) via bf16 WMMA, fp32 accumulate, f32 output.
// One wave per 16x16 C tile, K in steps of 32.
// ---------------------------------------------------------------------------
__global__ __launch_bounds__(32)
void gemm_bf16_kernel(const float* __restrict__ A, const float* __restrict__ B,
                      float* __restrict__ Cf, int K, int Nc)
{
    const int lane = threadIdx.x & 31;
    const int r16  = lane & 15;
    const int half = lane >> 4;
    const int m0   = blockIdx.y * 16;
    const int n0   = blockIdx.x * 16;

    v8f acc = {0.f,0.f,0.f,0.f,0.f,0.f,0.f,0.f};
    const float* arow = A + (size_t)(m0 + r16) * K;

    for (int k0 = 0; k0 < K; k0 += 32) {
        BF16x16 a, b;
#pragma unroll
        for (int r = 0; r < 8; ++r) {
            // A-tile: K-offset = (r>>2)*16 + half*8 + (r&3)*2 (ISA 16-bit A layout)
            const int ko = ((r >> 2) << 4) + half * 8 + ((r & 3) << 1);
            const float2 av = *(const float2*)(arow + k0 + ko);
            a.u[r] = pack2bf(av.x, av.y);
            // B-tile: same K-offset as A-tile (ISA 16-bit layout)
            const int kb = ((r >> 2) << 4) + half * 8 + ((r & 3) << 1);
            const float b0 = B[(size_t)(k0 + kb)     * Nc + n0 + r16];
            const float b1 = B[(size_t)(k0 + kb + 1) * Nc + n0 + r16];
            b.u[r] = pack2bf(b0, b1);
        }
        acc = __builtin_amdgcn_wmma_f32_16x16x32_bf16(false, a.v, false, b.v,
                                                      (short)0, acc, false, false);
    }
#pragma unroll
    for (int v = 0; v < 8; ++v)
        Cf[(size_t)(m0 + v + half * 8) * Nc + n0 + r16] = acc[v];
}

// ---------------------------------------------------------------------------
// Repack h[N,F] f32 into B-tile-ready bf16 layout:
//   hB[jb][ct][lane][r] dwords (jb = 32-row block, ct = 16-col tile), so the
//   attention loop fetches each WMMA B operand with two global_load_b128.
// ---------------------------------------------------------------------------
template<int F>
__global__ __launch_bounds__(32)
void repack_b_kernel(const float* __restrict__ h, unsigned int* __restrict__ hB)
{
    const int lane = threadIdx.x & 31;
    const int r16  = lane & 15;
    const int half = lane >> 4;
    const int jb = blockIdx.x;            // 32-row block of h
    const int ct = blockIdx.y;            // 16-col tile
    const int c  = ct * 16 + r16;
    BF16x16 b;
#pragma unroll
    for (int r = 0; r < 8; ++r) {
        const int k = jb * 32 + ((r >> 2) << 4) + half * 8 + ((r & 3) << 1);
        b.u[r] = pack2bf(h[(size_t)k * F + c], h[(size_t)(k + 1) * F + c]);
    }
    uint4* dst = (uint4*)(hB + ((size_t)(jb * (F / 16) + ct) * 32 + lane) * 8);
    dst[0] = b.q[0];
    dst[1] = b.q[1];
}

// ---------------------------------------------------------------------------
// Per-row projections f_self = h@a_self, f_neigh = h@a_neigh.
// One wave per row, lane-strided coalesced loads, shuffle reduction.
// ---------------------------------------------------------------------------
__global__ void fvec_kernel(const float* __restrict__ h,
                            const float* __restrict__ a_s,
                            const float* __restrict__ a_n,
                            float* __restrict__ fs, float* __restrict__ fn, int F)
{
    const int row  = (blockIdx.x * blockDim.x + threadIdx.x) >> 5;
    const int lane = threadIdx.x & 31;
    if (row >= NROWS) return;
    const float* hr = h + (size_t)row * F;
    float s = 0.f, n = 0.f;
    for (int k = lane; k < F; k += 32) {
        const float v = hr[k];
        s += v * a_s[k];
        n += v * a_n[k];
    }
#pragma unroll
    for (int d = 16; d >= 1; d >>= 1) {
        s += __shfl_xor(s, d, 32);
        n += __shfl_xor(n, d, 32);
    }
    if (lane == 0) { fs[row] = s; fn[row] = n; }
}

// ---------------------------------------------------------------------------
// Fused attention layer, flash-decoding style: one workgroup per 16-row
// block; JW waves each own a disjoint j (key) stripe and carry the FULL
// F-wide accumulator (NT = F/16 WMMA C-tiles). Online softmax state lives
// in registers (rm/rs are full-row after the xor-16 shuffle; C-layout
// correction factors come from 8 __shfl's). e/p computed exactly once per
// key tile -> no redundant VALU, no barriers in the stream loop. M/adj are
// streamed with NON-TEMPORAL loads (read-once, > L2 combined) so L2 keeps
// the reused operands (hB, fn). Partial (m, l, acc) merged once at the end
// in LDS (max-rescale + ds_add_f32). All NT WMMAs in a step share the A
// operand -> reuse-A hint on repeats.
// ---------------------------------------------------------------------------
template<int F, int JW>
__global__ __launch_bounds__(32 * JW)
void attn_layer_kernel(const float* __restrict__ Mm, const float* __restrict__ adj,
                       const float* __restrict__ fs, const float* __restrict__ fn,
                       const unsigned int* __restrict__ hB,   // repacked bf16 h
                       float* __restrict__ out)               // [N,F] f32
{
    constexpr int NT = F / 16;
    __shared__ float smw[JW][16];
    __shared__ float accb[16][F];
    __shared__ float lb[16];

    const int tid  = threadIdx.x;
    const int lane = tid & 31;
    const int wave = tid >> 5;
    const int r16  = lane & 15;
    const int half = lane >> 4;
    const int i0   = blockIdx.x * 16;
    const int row  = i0 + r16;

    const float fsv = fs[row];
    v8f acc[NT];
#pragma unroll
    for (int t = 0; t < NT; ++t) acc[t] = v8f{0.f,0.f,0.f,0.f,0.f,0.f,0.f,0.f};
    float m_run = -INFINITY, l_run = 0.f;

    const f4v* M4 = (const f4v*)(Mm  + (size_t)row * NROWS);
    const f4v* A4 = (const f4v*)(adj + (size_t)row * NROWS);

    for (int j0 = wave * 32; j0 < NROWS; j0 += JW * 32) {
        if (j0 + JW * 32 < NROWS) {   // stream hint -> global_prefetch_b8
            __builtin_prefetch((const float*)M4 + j0 + JW * 32, 0, 0);
            __builtin_prefetch((const float*)A4 + j0 + JW * 32, 0, 0);
        }
        // two contiguous 8-float runs per lane: K = half*8+t and 16+half*8+t
        const int q = (j0 >> 2) + half * 2;
        f4v mv[4] = { __builtin_nontemporal_load(M4 + q),
                      __builtin_nontemporal_load(M4 + q + 1),
                      __builtin_nontemporal_load(M4 + q + 4),
                      __builtin_nontemporal_load(M4 + q + 5) };
        f4v av[4] = { __builtin_nontemporal_load(A4 + q),
                      __builtin_nontemporal_load(A4 + q + 1),
                      __builtin_nontemporal_load(A4 + q + 4),
                      __builtin_nontemporal_load(A4 + q + 5) };
        const f4v* F4 = (const f4v*)(fn + j0);
        f4v fv[4] = { F4[half * 2], F4[half * 2 + 1],
                      F4[4 + half * 2], F4[4 + half * 2 + 1] };
        const float* mf = (const float*)mv;
        const float* af = (const float*)av;
        const float* ff = (const float*)fv;

        float e[16];
        float lm = -INFINITY;
#pragma unroll
        for (int t = 0; t < 16; ++t) {
            float v = (fsv + ff[t]) * mf[t];
            v = v > 0.f ? v : ALPHA * v;           // leaky_relu
            e[t] = af[t] > 0.f ? v : NEG_INF;      // adjacency mask
            lm = fmaxf(lm, e[t]);
        }
        const float rm = fmaxf(lm, __shfl_xor(lm, 16, 32));  // full-row step max
        const float nm = fmaxf(m_run, rm);
        const float corr = __expf(m_run - nm);
        m_run = nm;

        float ps = 0.f;
#pragma unroll
        for (int t = 0; t < 16; ++t) { e[t] = __expf(e[t] - nm); ps += e[t]; }
        const float rs = ps + __shfl_xor(ps, 16, 32);
        l_run = l_run * corr + rs;

        // p -> bf16 A-tile: reg r <- t pair at (r>>2)*8 + (r&3)*2
        BF16x16 pa;
#pragma unroll
        for (int r = 0; r < 8; ++r) {
            const int t0 = ((r >> 2) << 3) + ((r & 3) << 1);
            pa.u[r] = pack2bf(e[t0], e[t0 + 1]);
        }
        // correction factors for C-layout rows v+half*8 via shuffle
        float cs[8];
#pragma unroll
        for (int v = 0; v < 8; ++v) cs[v] = __shfl(corr, v + half * 8, 32);

        const unsigned int* tile = hB + ((size_t)(j0 >> 5) * NT * 32 + lane) * 8;
        {   // first tile: no reuse hint
            BF16x16 bb;
            bb.q[0] = ((const uint4*)tile)[0];
            bb.q[1] = ((const uint4*)tile)[1];
#pragma unroll
            for (int v = 0; v < 8; ++v) acc[0][v] *= cs[v];
            acc[0] = __builtin_amdgcn_wmma_f32_16x16x32_bf16(false, pa.v, false, bb.v,
                                                             (short)0, acc[0], false, false);
        }
#pragma unroll
        for (int t_ = 1; t_ < NT; ++t_) {   // same A operand -> reuse-A hint
            BF16x16 bb;
            const uint4* bp = (const uint4*)(tile + (size_t)t_ * 32 * 8);
            bb.q[0] = bp[0];
            bb.q[1] = bp[1];
#pragma unroll
            for (int v = 0; v < 8; ++v) acc[t_][v] *= cs[v];
            acc[t_] = __builtin_amdgcn_wmma_f32_16x16x32_bf16(false, pa.v, false, bb.v,
                                                              (short)0, acc[t_], true, false);
        }
    }

    // ---- merge JW partial softmax accumulators ----
    if (lane < 16) smw[wave][lane] = m_run;
    for (int idx = tid; idx < 16 * F; idx += 32 * JW) accb[idx / F][idx % F] = 0.f;
    if (tid < 16) lb[tid] = 0.f;
    __syncthreads();

    float mmax = -INFINITY;
#pragma unroll
    for (int w = 0; w < JW; ++w) mmax = fmaxf(mmax, smw[w][r16]);
    const float fac = __expf(m_run - mmax);
    if (lane < 16) atomicAdd(&lb[lane], l_run * fac);
    float fc[8];
#pragma unroll
    for (int v = 0; v < 8; ++v) fc[v] = __shfl(fac, v + half * 8, 32);
#pragma unroll
    for (int t_ = 0; t_ < NT; ++t_)
#pragma unroll
        for (int v = 0; v < 8; ++v)
            atomicAdd(&accb[v + half * 8][t_ * 16 + r16], acc[t_][v] * fc[v]);
    __syncthreads();

    for (int idx = tid; idx < 16 * F; idx += 32 * JW) {
        const int rr = idx / F, cc = idx % F;
        float val = accb[rr][cc] / lb[rr];
        val = val > 0.f ? val : (__expf(val) - 1.0f);   // elu
        out[(size_t)(i0 + rr) * F + cc] = val;
    }
}

// ---------------------------------------------------------------------------
// Row L2-normalize; emit z (f32, d_out tail) and zero-padded bf16 z [N,32]
// ---------------------------------------------------------------------------
__global__ void normalize_kernel(const float* __restrict__ h,
                                 float* __restrict__ zout,
                                 unsigned int* __restrict__ zbf)
{
    const int i = blockIdx.x * blockDim.x + threadIdx.x;
    if (i >= NROWS) return;
    float v[EMB], s = 0.f;
#pragma unroll
    for (int k = 0; k < EMB; ++k) { v[k] = h[(size_t)i * EMB + k]; s += v[k] * v[k]; }
    const float inv = 1.0f / fmaxf(sqrtf(s), 1e-12f);
    unsigned int pk[8];
#pragma unroll
    for (int k = 0; k < EMB; ++k) v[k] *= inv;
#pragma unroll
    for (int k = 0; k < 8; ++k) pk[k] = pack2bf(v[2 * k], v[2 * k + 1]);
#pragma unroll
    for (int k = 0; k < EMB; ++k) zout[(size_t)i * EMB + k] = v[k];
    uint4* dst = (uint4*)(zbf + (size_t)i * 16);
    dst[0] = uint4{pk[0], pk[1], pk[2], pk[3]};
    dst[1] = uint4{pk[4], pk[5], pk[6], pk[7]};
    dst[2] = uint4{0u, 0u, 0u, 0u};          // pad K 16..31 with zeros
    dst[3] = uint4{0u, 0u, 0u, 0u};
}

// ---------------------------------------------------------------------------
// A_pred = sigmoid(s - 1/s), s = z@z^T. One wave per 16x64 strip: 4 WMMAs
// sharing the A operand (reuse-A on repeats), K=16 zero-padded to 32.
// A_pred is written once and never read -> non-temporal stores.
// ---------------------------------------------------------------------------
__global__ __launch_bounds__(32)
void zzt_kernel(const unsigned int* __restrict__ zbf, float* __restrict__ Ap)
{
    const int lane = threadIdx.x & 31;
    const int r16  = lane & 15;
    const int half = lane >> 4;
    const int i0 = blockIdx.y * 16;
    const int j0 = blockIdx.x * 64;
    const uint4 z4 = {0u, 0u, 0u, 0u};

    const uint4* arow = (const uint4*)(zbf + (size_t)(i0 + r16) * 16);
    BF16x16 a;
    a.q[0] = arow[half];      // lanes: K 0..7 / 8..15
    a.q[1] = z4;              // K >= 16 padded zero

    v8f c[4];
#pragma unroll
    for (int t = 0; t < 4; ++t) {
        const uint4* brow = (const uint4*)(zbf + (size_t)(j0 + t * 16 + r16) * 16);
        BF16x16 b;
        b.q[0] = brow[half];            // lanes: K 0..7 / 8..15
        b.q[1] = z4;                    // K >= 16 padded zero
        v8f cz = {0.f,0.f,0.f,0.f,0.f,0.f,0.f,0.f};
        c[t] = (t == 0)
             ? __builtin_amdgcn_wmma_f32_16x16x32_bf16(false, a.v, false, b.v,
                                                       (short)0, cz, false, false)
             : __builtin_amdgcn_wmma_f32_16x16x32_bf16(false, a.v, false, b.v,
                                                       (short)0, cz, true, false);
    }
#pragma unroll
    for (int t = 0; t < 4; ++t)
#pragma unroll
        for (int v = 0; v < 8; ++v) {
            const float s = c[t][v];
            const float x = s - 1.0f / s;
            const float ap = 1.0f / (1.0f + __expf(-x));
            __builtin_nontemporal_store(
                ap, Ap + (size_t)(i0 + v + half * 8) * NROWS + j0 + t * 16 + r16);
        }
}

// ---------------------------------------------------------------------------
extern "C" void kernel_launch(void* const* d_in, const int* in_sizes, int n_in,
                              void* d_out, int out_size, void* d_ws, size_t ws_size,
                              hipStream_t stream)
{
    const float* x   = (const float*)d_in[0];   // [N,ATTR]
    const float* adj = (const float*)d_in[1];   // [N,N]
    const float* Mm  = (const float*)d_in[2];   // [N,N]
    const float* W1  = (const float*)d_in[3];   // [ATTR,HID]
    const float* as1 = (const float*)d_in[4];   // [HID]
    const float* an1 = (const float*)d_in[5];   // [HID]
    const float* W2  = (const float*)d_in[6];   // [HID,EMB]
    const float* as2 = (const float*)d_in[7];   // [EMB]
    const float* an2 = (const float*)d_in[8];   // [EMB]
    float* out = (float*)d_out;                 // A_pred [N,N] then z [N,EMB]

    // workspace carve-up
    char* ws = (char*)d_ws;
    size_t off = 0;
    auto take = [&](size_t bytes) { char* p = ws + off; off += (bytes + 255) & ~size_t(255); return p; };
    float*        h1f   = (float*)take((size_t)NROWS * HID * 4);
    unsigned int* h1B   = (unsigned int*)take((size_t)NROWS * HID * 2);
    float*        fs1   = (float*)take((size_t)NROWS * 4);
    float*        fn1   = (float*)take((size_t)NROWS * 4);
    float*        hout1 = (float*)take((size_t)NROWS * HID * 4);
    float*        h2f   = (float*)take((size_t)NROWS * EMB * 4);
    unsigned int* h2B   = (unsigned int*)take((size_t)NROWS * EMB * 2);
    float*        fs2   = (float*)take((size_t)NROWS * 4);
    float*        fn2   = (float*)take((size_t)NROWS * 4);
    float*        hout2 = (float*)take((size_t)NROWS * EMB * 4);
    unsigned int* zbf   = (unsigned int*)take((size_t)NROWS * 32 * 2);

    // ---- layer 1 ----
    gemm_bf16_kernel<<<dim3(HID / 16, NROWS / 16), 32, 0, stream>>>(x, W1, h1f, ATTR, HID);
    repack_b_kernel<HID><<<dim3(NROWS / 32, HID / 16), 32, 0, stream>>>(h1f, h1B);
    fvec_kernel<<<NROWS * 32 / 256, 256, 0, stream>>>(h1f, as1, an1, fs1, fn1, HID);
    attn_layer_kernel<HID, 8><<<NROWS / 16, 256, 0, stream>>>(Mm, adj, fs1, fn1, h1B, hout1);

    // ---- layer 2 ----
    gemm_bf16_kernel<<<dim3(EMB / 16, NROWS / 16), 32, 0, stream>>>(hout1, W2, h2f, HID, EMB);
    repack_b_kernel<EMB><<<dim3(NROWS / 32, EMB / 16), 32, 0, stream>>>(h2f, h2B);
    fvec_kernel<<<NROWS * 32 / 256, 256, 0, stream>>>(h2f, as2, an2, fs2, fn2, EMB);
    attn_layer_kernel<EMB, 8><<<NROWS / 16, 256, 0, stream>>>(Mm, adj, fs2, fn2, h2B, hout2);

    // ---- decode ----
    normalize_kernel<<<(NROWS + 255) / 256, 256, 0, stream>>>(hout2, out + (size_t)NROWS * NROWS, zbf);
    zzt_kernel<<<dim3(NROWS / 64, NROWS / 16), 32, 0, stream>>>(zbf, out);
}
